// BioWaveKAN_44684839747976
// MI455X (gfx1250) — matrix operand order repaired and checked by reading, not compiled, not run on hardware
//
#include <hip/hip_runtime.h>
#include <hip/hip_bf16.h>
#include <math.h>

// ---------------------------------------------------------------------------
// BioWaveKAN forward for MI455X (gfx1250): single fused bf16 WMMA GEMM
//   y = [wavelet(x) | x] (B x 4096 bf16)  @  [Ww | 0.3*Wb]^T (O x 4096 bf16)
// then deterministic BatchNorm (batch stats) epilogue.
// A-tile staging uses the Tensor Data Mover (tensor_load_to_lds) with the
// LDS pad feature reproducing the bank-padded stride.
// ---------------------------------------------------------------------------

typedef __attribute__((ext_vector_type(16))) __bf16 v16bf;
typedef __attribute__((ext_vector_type(8)))  float  v8f;
typedef __attribute__((ext_vector_type(4)))  unsigned int v4u;
typedef __attribute__((ext_vector_type(8)))  int v8i;
typedef __attribute__((ext_vector_type(4)))  int v4i;

#define IN_DIM   2048
#define OUT_DIM  2048
#define BATCH    4096
#define KDIM     4096          // concatenated K = 2*IN_DIM
#define BN_EPS   1e-5f

#define BM 128
#define BN 128
#define BK 32
#define SA 40                  // A-tile LDS row stride (bf16 elems), 80 B = 64 B data + 16 B pad
#define SW 136                 // W^T-tile LDS row stride (bf16 elems), 272 B

__device__ __forceinline__ unsigned short f2bf(float f) {
    union { float f; unsigned u; } v; v.f = f;
    unsigned u = v.u;
    unsigned r = u + 0x7FFFu + ((u >> 16) & 1u);   // round-to-nearest-even
    return (unsigned short)(r >> 16);
}

// ---- kernel 1: activation prep: act[b] = [wavelet(x[b]) | x[b]] in bf16 ----
__global__ void bwk_prep_act(const float* __restrict__ x,
                             const float* __restrict__ scale,
                             const float* __restrict__ translate,
                             unsigned short* __restrict__ act) {
    int idx = blockIdx.x * 256 + threadIdx.x;          // one float4 of x
    int b = idx / (IN_DIM / 4);
    int k = (idx % (IN_DIM / 4)) * 4;
    float4 x4 = ((const float4*)x)[idx];
    float4 s4 = *(const float4*)(scale + k);
    float4 t4 = *(const float4*)(translate + k);
    float xs[4] = {x4.x, x4.y, x4.z, x4.w};
    float ss[4] = {s4.x, s4.y, s4.z, s4.w};
    float ts[4] = {t4.x, t4.y, t4.z, t4.w};
    unsigned short w16[4], e16[4];
#pragma unroll
    for (int j = 0; j < 4; ++j) {
        float u = (xs[j] - ts[j]) / fmaxf(ss[j], 1e-3f);
        float wl = 0.75112554446494248f * __cosf(3.0f * u) * __expf(-0.5f * u * u);
        w16[j] = f2bf(wl);
        e16[j] = f2bf(xs[j]);
    }
    unsigned short* row = act + (size_t)b * KDIM;
    uint2 wpk = make_uint2((unsigned)w16[0] | ((unsigned)w16[1] << 16),
                           (unsigned)w16[2] | ((unsigned)w16[3] << 16));
    uint2 xpk = make_uint2((unsigned)e16[0] | ((unsigned)e16[1] << 16),
                           (unsigned)e16[2] | ((unsigned)e16[3] << 16));
    *(uint2*)(row + k)           = wpk;
    *(uint2*)(row + IN_DIM + k)  = xpk;
}

// ---- kernel 2: weight prep: wc[o] = [Ww[o] | 0.3*Wb[o]] in bf16 -----------
__global__ void bwk_prep_w(const float* __restrict__ ww,
                           const float* __restrict__ bw,
                           unsigned short* __restrict__ wc) {
    int idx = blockIdx.x * 256 + threadIdx.x;          // one float4
    int o = idx / (IN_DIM / 4);
    int k = (idx % (IN_DIM / 4)) * 4;
    float4 w4 = ((const float4*)ww)[idx];
    float4 b4 = ((const float4*)bw)[idx];
    unsigned short a16[4], c16[4];
    float wsv[4] = {w4.x, w4.y, w4.z, w4.w};
    float bsv[4] = {b4.x, b4.y, b4.z, b4.w};
#pragma unroll
    for (int j = 0; j < 4; ++j) {
        a16[j] = f2bf(wsv[j]);
        c16[j] = f2bf(0.3f * bsv[j]);
    }
    unsigned short* row = wc + (size_t)o * KDIM;
    *(uint2*)(row + k)          = make_uint2((unsigned)a16[0] | ((unsigned)a16[1] << 16),
                                             (unsigned)a16[2] | ((unsigned)a16[3] << 16));
    *(uint2*)(row + IN_DIM + k) = make_uint2((unsigned)c16[0] | ((unsigned)c16[1] << 16),
                                             (unsigned)c16[2] | ((unsigned)c16[3] << 16));
}

// ---- kernel 3: bf16 WMMA GEMM: y(4096x2048) = act(4096x4096) * wc^T -------
__global__ void __launch_bounds__(256)
bwk_gemm_bf16(const unsigned short* __restrict__ act,
              const unsigned short* __restrict__ wc,
              float* __restrict__ y) {
    __shared__ unsigned short As[BM * SA];   // [m][k], K-contiguous, TDM-filled
    __shared__ unsigned short Ws[BK * SW];   // [k][n], transposed W tile

    const int t    = threadIdx.x;
    const int lane = t & 31;                 // wave32
    const int wid  = t >> 5;                 // 8 waves
    const int m0   = blockIdx.y * BM;
    const int n0   = blockIdx.x * BN;

    // --- TDM descriptor group 1 (iteration-invariant):
    //   data_size=2B; pad_enable; pad_interval=16 dwords (code 3);
    //   pad_amount=4 dwords (code 3) -> LDS row stride 80 B;
    //   tensor_dim0 = KDIM, tensor_dim1 = BATCH;
    //   tile_dim0 = 32 elems (64 B rows), tile_dim1 = 128 rows;
    //   tensor_dim0_stride = KDIM elems.
    v8i g1;
    g1[0] = (int)((1u << 16) | (1u << 20) | (3u << 22) | (3u << 25));
    g1[1] = (int)((unsigned)KDIM << 16);     // tensor_dim0[15:0] @ bits 63:48
    g1[2] = (int)((unsigned)BATCH << 16);    // dim0 hi=0 | tensor_dim1[15:0]
    g1[3] = (int)((unsigned)BK << 16);       // dim1 hi=0 | tile_dim0=32
    g1[4] = BM;                              // tile_dim1=128 | tile_dim2=0
    g1[5] = KDIM;                            // tensor_dim0_stride lo32
    g1[6] = 0;                               // stride hi | dim1_stride lo
    g1[7] = 0;
    const v4i gz4 = {0, 0, 0, 0};
    const v8i gz8 = {0, 0, 0, 0, 0, 0, 0, 0};
    const unsigned lds_a = (unsigned)(uintptr_t)(&As[0]);
    const unsigned long long act_base = (unsigned long long)(uintptr_t)act
                                      + (unsigned long long)m0 * KDIM * 2ull;

    const v8f vzero = {};
    v8f acc[8];
#pragma unroll
    for (int i = 0; i < 8; ++i) acc[i] = vzero;

    for (int k0 = 0; k0 < KDIM; k0 += BK) {
        // --- stage A tile via Tensor Data Mover (one wave issues the DMA)
        if (wid == 0) {
            unsigned long long ga = act_base + (unsigned long long)k0 * 2ull;
            v4u g0;
            g0[0] = 1u;                                          // count=1
            g0[1] = lds_a;                                       // LDS dest
            g0[2] = (unsigned)ga;                                // addr lo
            g0[3] = (unsigned)((ga >> 32) & 0x1FFFFFFu) | (2u << 30); // addr hi | type=2
            __builtin_amdgcn_tensor_load_to_lds(g0, g1, gz4, gz4, gz8, 0);
        }
        // --- stage W tile transposed (all threads): Ws[k][n] = wc[n0+n][k0+k]
#pragma unroll
        for (int ch = 0; ch < 2; ++ch) {
            int c  = t + ch * 256;
            int n  = c >> 2;
            int kc = (c & 3) * 8;
            uint4 v = *(const uint4*)(wc + (size_t)(n0 + n) * KDIM + k0 + kc);
            unsigned short e[8] = {
                (unsigned short)(v.x), (unsigned short)(v.x >> 16),
                (unsigned short)(v.y), (unsigned short)(v.y >> 16),
                (unsigned short)(v.z), (unsigned short)(v.z >> 16),
                (unsigned short)(v.w), (unsigned short)(v.w >> 16)};
#pragma unroll
            for (int j = 0; j < 8; ++j)
                Ws[(kc + j) * SW + n] = e[j];
        }
        // prefetch next W K-tile while we compute (global_prefetch_b8)
        if (k0 + BK < KDIM) {
            __builtin_prefetch(wc + (size_t)(n0 + (t >> 1)) * KDIM + k0 + BK, 0, 1);
        }
        if (wid == 0) {
            __builtin_amdgcn_s_wait_tensorcnt(0);
        }
        __syncthreads();

        // --- A fragment (ISA 16-bit A 16x32 layout):
        //     lane<16: K 0-7 then 16-23 ; lane>=16: K 8-15 then 24-31
        union U16 { uint4 q[2]; v16bf v; };
        U16 af;
        int am = wid * 16 + (lane & 15);
        int ak = (lane & 16) ? 8 : 0;
        af.q[0] = *(const uint4*)(As + am * SA + ak);
        af.q[1] = *(const uint4*)(As + am * SA + ak + 16);

        // --- load all 8 B fragments first (one LDS clause), then 8 WMMAs
        U16 bfr[8];
#pragma unroll
        for (int nt = 0; nt < 8; ++nt) {
#pragma unroll
            for (int s = 0; s < 16; ++s) {
                int kk = (s < 8) ? ((lane >> 4) * 8 + s) : (16 + (lane >> 4) * 8 + (s - 8));
                bfr[nt].v[s] = __builtin_bit_cast(__bf16, Ws[kk * SW + nt * 16 + (lane & 15)]);
            }
        }
#pragma unroll
        for (int nt = 0; nt < 8; ++nt) {
            acc[nt] = __builtin_amdgcn_wmma_f32_16x16x32_bf16(
                false, af.v, false, bfr[nt].v, (short)0, acc[nt], false, false);
        }
        __syncthreads();
    }

    // --- store: C/D layout: VGPR j -> M = j (+8 for lanes 16-31), N = lane&15
    int mrow = m0 + wid * 16 + ((lane >> 4) ? 8 : 0);
    int ncol = n0 + (lane & 15);
#pragma unroll
    for (int nt = 0; nt < 8; ++nt) {
#pragma unroll
        for (int j = 0; j < 8; ++j)
            y[(size_t)(mrow + j) * OUT_DIM + ncol + nt * 16] = acc[nt][j];
    }
}

// ---- kernel 4: per-column partial sums over 128-row chunks (deterministic)
__global__ void bwk_colreduce(const float* __restrict__ y,
                              float* __restrict__ psum,
                              float* __restrict__ psq) {
    int c     = blockIdx.x * 256 + threadIdx.x;
    int chunk = blockIdx.y;                         // 0..31
    int r0    = chunk * (BATCH / 32);
    float s = 0.f, q = 0.f;
    for (int r = 0; r < BATCH / 32; ++r) {
        float v = y[(size_t)(r0 + r) * OUT_DIM + c];
        s += v;
        q += v * v;
    }
    psum[chunk * OUT_DIM + c] = s;
    psq [chunk * OUT_DIM + c] = q;
}

// ---- kernel 5: finalize BN affine coefficients ----------------------------
__global__ void bwk_bn_finalize(const float* __restrict__ psum,
                                const float* __restrict__ psq,
                                const float* __restrict__ gamma,
                                const float* __restrict__ beta,
                                float* __restrict__ ab) {
    int c = blockIdx.x * 256 + threadIdx.x;
    float s = 0.f, q = 0.f;
    for (int ch = 0; ch < 32; ++ch) {
        s += psum[ch * OUT_DIM + c];
        q += psq [ch * OUT_DIM + c];
    }
    float mean = s * (1.0f / BATCH);
    float var  = q * (1.0f / BATCH) - mean * mean;   // biased variance
    float inv  = rsqrtf(fmaxf(var, 0.0f) + BN_EPS);
    float A = gamma[c] * inv;
    ab[c]           = A;
    ab[OUT_DIM + c] = beta[c] - A * mean;
}

// ---- kernel 6: apply BN affine --------------------------------------------
__global__ void bwk_bn_apply(const float* __restrict__ y,
                             const float* __restrict__ ab,
                             float* __restrict__ out) {
    int idx = blockIdx.x * 256 + threadIdx.x;       // one float4 of y
    int c = (idx % (OUT_DIM / 4)) * 4;
    float4 v = ((const float4*)y)[idx];
    float4 A = *(const float4*)(ab + c);
    float4 B = *(const float4*)(ab + OUT_DIM + c);
    float4 o;
    o.x = v.x * A.x + B.x;
    o.y = v.y * A.y + B.y;
    o.z = v.z * A.z + B.z;
    o.w = v.w * A.w + B.w;
    ((float4*)out)[idx] = o;
}

extern "C" void kernel_launch(void* const* d_in, const int* in_sizes, int n_in,
                              void* d_out, int out_size, void* d_ws, size_t ws_size,
                              hipStream_t stream) {
    const float* x         = (const float*)d_in[0];
    const float* scale     = (const float*)d_in[1];
    const float* translate = (const float*)d_in[2];
    const float* ww        = (const float*)d_in[3];
    const float* bw        = (const float*)d_in[4];
    const float* gamma     = (const float*)d_in[5];
    const float* beta      = (const float*)d_in[6];
    float* out = (float*)d_out;

    char* ws = (char*)d_ws;
    unsigned short* act = (unsigned short*)(ws);                 // 32 MB
    unsigned short* wc  = (unsigned short*)(ws + (size_t)33554432);  // 16 MB
    float* y    = (float*)(ws + (size_t)50331648);               // 32 MB
    float* psum = (float*)(ws + (size_t)83886080);               // 256 KB
    float* psq  = (float*)(ws + (size_t)84148224);               // 256 KB
    float* ab   = (float*)(ws + (size_t)84410368);               // 16 KB

    bwk_prep_act<<<(BATCH * IN_DIM / 4) / 256, 256, 0, stream>>>(x, scale, translate, act);
    bwk_prep_w  <<<(OUT_DIM * IN_DIM / 4) / 256, 256, 0, stream>>>(ww, bw, wc);
    bwk_gemm_bf16<<<dim3(OUT_DIM / BN, BATCH / BM), 256, 0, stream>>>(act, wc, y);
    bwk_colreduce<<<dim3(OUT_DIM / 256, 32), 256, 0, stream>>>(y, psum, psq);
    bwk_bn_finalize<<<OUT_DIM / 256, 256, 0, stream>>>(psum, psq, gamma, beta, ab);
    bwk_bn_apply<<<(BATCH * OUT_DIM / 4) / 256, 256, 0, stream>>>(y, ab, out);
}
